// LSTMClassifier_32710470927037
// MI455X (gfx1250) — matrix operand order repaired and checked by reading, not compiled, not run on hardware
//
#include <hip/hip_runtime.h>
#include <hip/hip_bf16.h>

#define B_   1024
#define T_   512
#define I_   128
#define H_   64
#define E_   128
#define C_   100
#define HP   80     // padded LDS row stride (bf16 elems) for h buffers: 160B -> bank spread, keeps 16B alignment

typedef __attribute__((ext_vector_type(16))) __bf16 v16bf;
typedef __attribute__((ext_vector_type(8)))  __bf16 v8bf;
typedef __attribute__((ext_vector_type(8)))  float  v8f;
typedef __attribute__((ext_vector_type(4)))  float  f4;

// D = A*B + C, bf16 inputs, f32 accumulate (CDNA5 dense WMMA)
#define WMMA_BF16(a, b, c) \
  __builtin_amdgcn_wmma_f32_16x16x32_bf16(false, (a), false, (b), (short)0, (c), false, false)

// Load 16 consecutive f32 -> bf16 vector (WMMA B-operand lane chunk: elems = K ascending)
static __device__ __forceinline__ v16bf cvt_b16(const float* __restrict__ p) {
  const f4* q = (const f4*)p;
  f4 a = q[0], b = q[1], c = q[2], d = q[3];
  v16bf r;
#pragma unroll
  for (int i = 0; i < 4; ++i) {
    r[i]      = (__bf16)a[i];
    r[i + 4]  = (__bf16)b[i];
    r[i + 8]  = (__bf16)c[i];
    r[i + 12] = (__bf16)d[i];
  }
  return r;
}

// A-operand tile chunking: elems 0..7 = row[k1..k1+8), elems 8..15 = row[k2..k2+8)  (f32 source)
static __device__ __forceinline__ v16bf loadA_g(const float* __restrict__ row, int k1, int k2) {
  const f4* p1 = (const f4*)(row + k1);
  const f4* p2 = (const f4*)(row + k2);
  f4 a = p1[0], b = p1[1], c = p2[0], d = p2[1];
  v16bf r;
#pragma unroll
  for (int i = 0; i < 4; ++i) {
    r[i]      = (__bf16)a[i];
    r[i + 4]  = (__bf16)b[i];
    r[i + 8]  = (__bf16)c[i];
    r[i + 12] = (__bf16)d[i];
  }
  return r;
}

// Same chunking from an LDS bf16 h-buffer (row stride HP)
static __device__ __forceinline__ v16bf loadA_lds(const __bf16* __restrict__ h, int m, int k1, int k2) {
  v8bf lo = *(const v8bf*)(h + m * HP + k1);
  v8bf hi = *(const v8bf*)(h + m * HP + k2);
  v16bf r;
#pragma unroll
  for (int i = 0; i < 8; ++i) { r[i] = lo[i]; r[i + 8] = hi[i]; }
  return r;
}

static __device__ __forceinline__ v8f vsig(v8f x) {
  v8f r;
#pragma unroll
  for (int i = 0; i < 8; ++i) r[i] = 1.f / (1.f + __expf(-x[i]));
  return r;
}
static __device__ __forceinline__ v8f vtanh(v8f x) {
  v8f r;
#pragma unroll
  for (int i = 0; i < 8; ++i) r[i] = tanhf(x[i]);
  return r;
}

__global__ __launch_bounds__(128, 1)
void lstm2_fused_head(const float* __restrict__ x,
                      const float* __restrict__ Wih0, const float* __restrict__ Whh0,
                      const float* __restrict__ bih0, const float* __restrict__ bhh0,
                      const float* __restrict__ Wih1, const float* __restrict__ Whh1,
                      const float* __restrict__ bih1, const float* __restrict__ bhh1,
                      const float* __restrict__ Wproj, const float* __restrict__ bproj,
                      const float* __restrict__ Wfc1,  const float* __restrict__ bfc1,
                      const float* __restrict__ Wfc2,  const float* __restrict__ bfc2,
                      float* __restrict__ out)
{
  __shared__ __align__(16) __bf16 h0s[16 * HP];   // layer0 hidden state tile, bf16
  __shared__ __align__(16) __bf16 h1s[16 * HP];   // layer1 hidden state tile, bf16
  __shared__ float embs[16 * E_];
  __shared__ float hids[16 * E_];

  const int tid  = threadIdx.x;
  const int lane = tid & 31;
  const int wv   = tid >> 5;      // wave 0..3: owns gate-columns [16*wv, 16*wv+16) of each gate
  const int hlf  = lane >> 4;     // lane-half selects K chunks / D row-half
  const int nl   = lane & 15;     // row (A/M) or column (B/D) index within tile
  const int b0   = blockIdx.x * 16;

  // ---- register-resident WMMA B operands (weights), k = 16*half + s per lane, same order as A ----
  v16bf Bx0[4][4], Bh0[4][2], Bx1[4][2], Bh1[4][2];
  float bias0[4], bias1[4];
#pragma unroll
  for (int g = 0; g < 4; ++g) {
    const int n = g * H_ + wv * 16 + nl;  // W row == gate output column
    bias0[g] = bih0[n] + bhh0[n];
    bias1[g] = bih1[n] + bhh1[n];
#pragma unroll
    for (int kt = 0; kt < 4; ++kt)
      Bx0[g][kt] = cvt_b16(Wih0 + n * I_ + kt * 32 + hlf * 16);
#pragma unroll
    for (int kt = 0; kt < 2; ++kt) {
      Bh0[g][kt] = cvt_b16(Whh0 + n * H_ + kt * 32 + hlf * 16);
      Bx1[g][kt] = cvt_b16(Wih1 + n * H_ + kt * 32 + hlf * 16);
      Bh1[g][kt] = cvt_b16(Whh1 + n * H_ + kt * 32 + hlf * 16);
    }
  }

  for (int i = tid; i < 16 * HP; i += 128) { h0s[i] = (__bf16)0.f; h1s[i] = (__bf16)0.f; }
  __syncthreads();

  v8f c0, c1;
#pragma unroll
  for (int i = 0; i < 8; ++i) { c0[i] = 0.f; c1[i] = 0.f; }

  const float* xrow = x + (size_t)(b0 + nl) * T_ * I_;   // this lane's batch row
  const int kc   = hlf * 16;        // A chunk base within a 32-wide K tile
  const int dcol = wv * 16 + nl;    // D/state column owned by this lane

  for (int t = 0; t < T_; ++t) {
    if (t + 1 < T_) __builtin_prefetch(xrow + I_, 0, 1);   // global_prefetch for next step

    v8f acc[4];

    // ================= layer 0 =================
#pragma unroll
    for (int g = 0; g < 4; ++g) {
#pragma unroll
      for (int i = 0; i < 8; ++i) acc[g][i] = bias0[g];
    }
#pragma unroll
    for (int kt = 0; kt < 4; ++kt) {            // x_t part, K=128
      v16bf a = loadA_g(xrow, kt * 32 + kc, kt * 32 + kc + 8);
#pragma unroll
      for (int g = 0; g < 4; ++g) acc[g] = WMMA_BF16(a, Bx0[g][kt], acc[g]);
    }
#pragma unroll
    for (int kt = 0; kt < 2; ++kt) {            // h(t-1) part, K=64
      v16bf a = loadA_lds(h0s, nl, kt * 32 + kc, kt * 32 + kc + 8);
#pragma unroll
      for (int g = 0; g < 4; ++g) acc[g] = WMMA_BF16(a, Bh0[g][kt], acc[g]);
    }
    __syncthreads();                            // all waves done reading h0(t-1)
    {
      v8f ig = vsig(acc[0]), fg = vsig(acc[1]), gg = vtanh(acc[2]), og = vsig(acc[3]);
      c0 = fg * c0 + ig * gg;
      v8f h = og * vtanh(c0);
#pragma unroll
      for (int r = 0; r < 8; ++r) h0s[(hlf * 8 + r) * HP + dcol] = (__bf16)h[r];
    }
    __syncthreads();                            // h0(t) visible

    // ================= layer 1 =================
#pragma unroll
    for (int g = 0; g < 4; ++g) {
#pragma unroll
      for (int i = 0; i < 8; ++i) acc[g][i] = bias1[g];
    }
#pragma unroll
    for (int kt = 0; kt < 2; ++kt) {            // input = h0(t)
      v16bf a = loadA_lds(h0s, nl, kt * 32 + kc, kt * 32 + kc + 8);
#pragma unroll
      for (int g = 0; g < 4; ++g) acc[g] = WMMA_BF16(a, Bx1[g][kt], acc[g]);
    }
#pragma unroll
    for (int kt = 0; kt < 2; ++kt) {            // h1(t-1)
      v16bf a = loadA_lds(h1s, nl, kt * 32 + kc, kt * 32 + kc + 8);
#pragma unroll
      for (int g = 0; g < 4; ++g) acc[g] = WMMA_BF16(a, Bh1[g][kt], acc[g]);
    }
    __syncthreads();                            // all waves done reading h1(t-1)
    {
      v8f ig = vsig(acc[0]), fg = vsig(acc[1]), gg = vtanh(acc[2]), og = vsig(acc[3]);
      c1 = fg * c1 + ig * gg;
      v8f h = og * vtanh(c1);
#pragma unroll
      for (int r = 0; r < 8; ++r) h1s[(hlf * 8 + r) * HP + dcol] = (__bf16)h[r];
    }
    __syncthreads();

    xrow += I_;
  }

  // ================= head MLP on h1(T-1) (tiny; scalar) =================
  {
    const int e = tid;                          // 0..127
    const float* wrow = Wproj + e * H_;
    const float bb = bproj[e];
#pragma unroll 1
    for (int m = 0; m < 16; ++m) {
      float s = bb;
      for (int k = 0; k < H_; ++k) s += (float)h1s[m * HP + k] * wrow[k];
      embs[m * E_ + e] = fmaxf(s, 0.f);
    }
  }
  __syncthreads();
  {
    const int e = tid;
    const float* wrow = Wfc1 + e * E_;
    const float bb = bfc1[e];
#pragma unroll 1
    for (int m = 0; m < 16; ++m) {
      float s = bb;
      for (int k = 0; k < E_; ++k) s += embs[m * E_ + k] * wrow[k];
      hids[m * E_ + e] = fmaxf(s, 0.f);
    }
  }
  __syncthreads();
  if (tid < C_) {
    const float* wrow = Wfc2 + tid * E_;
    const float bb = bfc2[tid];
#pragma unroll 1
    for (int m = 0; m < 16; ++m) {
      float s = bb;
      for (int k = 0; k < E_; ++k) s += hids[m * E_ + k] * wrow[k];
      out[(size_t)(b0 + m) * C_ + tid] = s;
    }
  }
}

extern "C" void kernel_launch(void* const* d_in, const int* in_sizes, int n_in,
                              void* d_out, int out_size, void* d_ws, size_t ws_size,
                              hipStream_t stream) {
  (void)in_sizes; (void)n_in; (void)out_size; (void)d_ws; (void)ws_size;
  const float* x     = (const float*)d_in[0];
  const float* Wih0  = (const float*)d_in[1];
  const float* Whh0  = (const float*)d_in[2];
  const float* bih0  = (const float*)d_in[3];
  const float* bhh0  = (const float*)d_in[4];
  const float* Wih1  = (const float*)d_in[5];
  const float* Whh1  = (const float*)d_in[6];
  const float* bih1  = (const float*)d_in[7];
  const float* bhh1  = (const float*)d_in[8];
  const float* Wproj = (const float*)d_in[9];
  const float* bproj = (const float*)d_in[10];
  const float* Wfc1  = (const float*)d_in[11];
  const float* bfc1  = (const float*)d_in[12];
  const float* Wfc2  = (const float*)d_in[13];
  const float* bfc2  = (const float*)d_in[14];
  float* out = (float*)d_out;

  lstm2_fused_head<<<dim3(B_ / 16), dim3(128), 0, stream>>>(
      x, Wih0, Whh0, bih0, bhh0, Wih1, Whh1, bih1, bhh1,
      Wproj, bproj, Wfc1, bfc1, Wfc2, bfc2, out);
}
